// Qnet_63221918597221
// MI455X (gfx1250) — matrix operand order repaired and checked by reading, not compiled, not run on hardware
//
#include <hip/hip_runtime.h>

// ---------------------------------------------------------------------------
// Variational GRU for MI455X (gfx1250, wave32, WMMA + async-to-LDS).
//   B=32, W=256, IN=4096, H=1024, Z=D=100
//   1. Convert W_ih / W_hh to bf16 (hardware v_cvt casts).
//   2. xW = x @ W_ih^T + b_ih  via v_wmma_f32_16x16x32_bf16 (bandwidth bound;
//      A operand converted f32->bf16 in-register with hw casts).
//   3. 256 sequential steps: h_{t-1} staged block-wide into LDS with
//      global_load_async_to_lds_b128 (ASYNCcnt), GEMM(h @ W_hh^T) with A from
//      LDS (ds_load_b128) fused with GRU gates; then a 1-workgroup heads
//      kernel (dense/mu/sigma/planar flow, fp32 scalar).
// ---------------------------------------------------------------------------

typedef __attribute__((ext_vector_type(16))) __bf16 v16bf;
typedef __attribute__((ext_vector_type(8)))  __bf16 v8bf;
typedef __attribute__((ext_vector_type(2)))  __bf16 v2bf;
typedef __attribute__((ext_vector_type(8)))  float  v8f;

#define B_SZ 32
#define W_T  256
#define K_IN 4096
#define H_D  1024
#define Z_D  100
#define D_D  100
#define G3H  3072   // 3*H

// ---------------- helpers ----------------

__device__ __forceinline__ float sigmoidf_(float x) {
  return 1.0f / (1.0f + __expf(-x));
}

__device__ __forceinline__ float tanhf_(float x) {
  float t  = fminf(fmaxf(x, -15.0f), 15.0f);
  float e2 = __expf(2.0f * t);
  return (e2 - 1.0f) / (e2 + 1.0f);
}

__device__ __forceinline__ v8f wmma_bf16(v16bf a, v16bf b, v8f c) {
  // (neg_a, A, neg_b, B, c_mod, C, reuse_a, reuse_b)
  return __builtin_amdgcn_wmma_f32_16x16x32_bf16(false, a, false, b,
                                                 (short)0, c, false, false);
}

// A tile 16x32 bf16, ISA layout: lanes 0-15 -> rows, K {0..7,16..23};
// lanes 16-31 -> K {8..15,24..31}.
__device__ __forceinline__ v16bf load_a_bf16(const __bf16* __restrict__ base,
                                             int ld, int row, int k0, int half) {
  const __bf16* p = base + (size_t)row * ld + k0 + half * 8;
  v8bf lo = *(const v8bf*)p;
  v8bf hi = *(const v8bf*)(p + 16);
  v16bf a;
#pragma unroll
  for (int i = 0; i < 8; ++i) { a[i] = lo[i]; a[8 + i] = hi[i]; }
  return a;
}

// A tile from fp32 source, converting to bf16 in-register (hw v_cvt).
__device__ __forceinline__ v16bf load_a_f32cvt(const float* __restrict__ rowp,
                                               int k0, int half) {
  const float* p = rowp + k0 + half * 8;
  float4 lo0 = *(const float4*)(p);
  float4 lo1 = *(const float4*)(p + 4);
  float4 hi0 = *(const float4*)(p + 16);
  float4 hi1 = *(const float4*)(p + 20);
  v16bf a;
  a[0]  = (__bf16)lo0.x; a[1]  = (__bf16)lo0.y; a[2]  = (__bf16)lo0.z; a[3]  = (__bf16)lo0.w;
  a[4]  = (__bf16)lo1.x; a[5]  = (__bf16)lo1.y; a[6]  = (__bf16)lo1.z; a[7]  = (__bf16)lo1.w;
  a[8]  = (__bf16)hi0.x; a[9]  = (__bf16)hi0.y; a[10] = (__bf16)hi0.z; a[11] = (__bf16)hi0.w;
  a[12] = (__bf16)hi1.x; a[13] = (__bf16)hi1.y; a[14] = (__bf16)hi1.z; a[15] = (__bf16)hi1.w;
  return a;
}

// B tile 32x16 bf16 from an [N][K] row-major matrix: each lane holds one
// column (lane&15), K {k0+half*8..+7, k0+16+half*8..+7} (matches A layout).
__device__ __forceinline__ v16bf load_b_bf16(const __bf16* __restrict__ base,
                                             int ld, int col, int k0, int half) {
  const __bf16* p = base + (size_t)col * ld + k0 + half * 8;
  v8bf lo = *(const v8bf*)p;
  v8bf hi = *(const v8bf*)(p + 16);
  v16bf b;
#pragma unroll
  for (int i = 0; i < 8; ++i) { b[i] = lo[i]; b[8 + i] = hi[i]; }
  return b;
}

// ---------------- kernels ----------------

__global__ void cvt_f32_to_bf16_k(const float* __restrict__ src,
                                  v2bf* __restrict__ dst, int n2) {
  int i = blockIdx.x * blockDim.x + threadIdx.x;
  if (i < n2) {
    float2 f = ((const float2*)src)[i];
    v2bf o; o[0] = (__bf16)f.x; o[1] = (__bf16)f.y;
    dst[i] = o;
  }
}

__global__ void init_state_k(float* __restrict__ hf, __bf16* __restrict__ hb,
                             float* __restrict__ z) {
  int i = blockIdx.x * blockDim.x + threadIdx.x;
  if (i < B_SZ * H_D) { hf[i] = 0.0f; hb[i] = (__bf16)0.0f; }
  if (i < B_SZ * Z_D) z[i] = 0.0f;
}

// xW[m][g] = sum_k x[b][w][k] * W_ih[g][k] + b_ih[g],  m = w*B + b.
// One wave -> 16x64 output strip (4 WMMA tiles sharing the A operand).
__global__ void __launch_bounds__(256)
input_proj_wmma(const float* __restrict__ x,       // [B, W, K_IN] fp32
                const __bf16* __restrict__ wih,    // [G3H, K_IN] bf16
                const float* __restrict__ b_ih,    // [G3H]
                float* __restrict__ xW)            // [W*B, G3H] fp32
{
  const int lane = threadIdx.x & 31;
  const int wave = threadIdx.x >> 5;
  const int unit = blockIdx.x * 8 + wave;       // 24576 units
  const int mt   = unit & 511;                  // 512 M tiles
  const int ng   = unit >> 9;                   // 48 N groups of 64 cols
  const int m0   = mt * 16;
  const int n0   = ng * 64;
  const int half = lane >> 4;

  const int m = m0 + (lane & 15);               // this lane's A row (w*B+b)
  const float* rowp = x + ((size_t)(m & (B_SZ - 1)) * W_T + (m >> 5)) * K_IN;
  const int ncol = n0 + (lane & 15);

  v8f a0 = {0,0,0,0,0,0,0,0}, a1 = {0,0,0,0,0,0,0,0};
  v8f a2 = {0,0,0,0,0,0,0,0}, a3 = {0,0,0,0,0,0,0,0};

  for (int k0 = 0; k0 < K_IN; k0 += 32) {
    v16bf a  = load_a_f32cvt(rowp, k0, half);
    v16bf b0 = load_b_bf16(wih, K_IN, ncol,      k0, half);
    v16bf b1 = load_b_bf16(wih, K_IN, ncol + 16, k0, half);
    v16bf b2 = load_b_bf16(wih, K_IN, ncol + 32, k0, half);
    v16bf b3 = load_b_bf16(wih, K_IN, ncol + 48, k0, half);
    a0 = wmma_bf16(a, b0, a0);
    a1 = wmma_bf16(a, b1, a1);
    a2 = wmma_bf16(a, b2, a2);
    a3 = wmma_bf16(a, b3, a3);
  }

  const int mrb = m0 + half * 8;                 // C/D layout rows
  const float bi0 = b_ih[ncol], bi1 = b_ih[ncol + 16];
  const float bi2 = b_ih[ncol + 32], bi3 = b_ih[ncol + 48];
#pragma unroll
  for (int i = 0; i < 8; ++i) {
    size_t r = (size_t)(mrb + i) * G3H;
    xW[r + ncol]      = a0[i] + bi0;
    xW[r + ncol + 16] = a1[i] + bi1;
    xW[r + ncol + 32] = a2[i] + bi2;
    xW[r + ncol + 48] = a3[i] + bi3;
  }
}

// One GRU timestep: gh = h @ W_hh^T (WMMA) fused with gate nonlinearities.
// h_{t-1} (64KB bf16) is staged block-wide into LDS with async-to-LDS loads,
// then A tiles come from LDS. 128 wave-units: 2 M-tiles x 64 N-tiles.
__global__ void __launch_bounds__(256)
gru_step_wmma(const __bf16* __restrict__ h_b_in,   // [B,H] bf16 (h_{t-1})
              const __bf16* __restrict__ whh,      // [G3H,H] bf16
              const float* __restrict__ b_hh,      // [G3H]
              const float* __restrict__ xw_t,      // [B,G3H] slice for step w
              const float* __restrict__ h_in,      // [B,H] fp32 (h_{t-1})
              float* __restrict__ h_out,           // [B,H] fp32 (h_t)
              __bf16* __restrict__ h_b_out)        // [B,H] bf16 (h_t)
{
  __shared__ __align__(16) __bf16 hsh[B_SZ * H_D];   // 64 KB

  // ---- stage h_{t-1} into LDS: 256 lanes x 16B x 16 iters = 64KB ----
  {
    const unsigned lbase = (unsigned)(size_t)&hsh[0];   // LDS byte offset
#pragma unroll
    for (int i = 0; i < 16; ++i) {
      const unsigned byte_off = (unsigned)(i * 256 + threadIdx.x) * 16u;
      const unsigned loff = lbase + byte_off;
      const char* gp = (const char*)h_b_in + byte_off;
      asm volatile("global_load_async_to_lds_b128 %0, %1, off"
                   :: "v"(loff), "v"(gp) : "memory");
    }
    asm volatile("s_wait_asynccnt 0x0" ::: "memory");
  }
  __syncthreads();

  const int lane = threadIdx.x & 31;
  const int wave = threadIdx.x >> 5;
  const int unit = blockIdx.x * 8 + wave;        // 0..127
  const int m0   = (unit & 1) * 16;
  const int n0   = (unit >> 1) * 16;
  const int half = lane >> 4;
  const int row  = m0 + (lane & 15);
  const int ncol = n0 + (lane & 15);

  v8f ar = {0,0,0,0,0,0,0,0};
  v8f az = {0,0,0,0,0,0,0,0};
  v8f an = {0,0,0,0,0,0,0,0};

  for (int k0 = 0; k0 < H_D; k0 += 32) {
    // A tile from LDS (ds_load_b128): ISA A-operand layout.
    v16bf a;
    {
      const __bf16* p = &hsh[row * H_D + k0 + half * 8];
      v8bf lo = *(const v8bf*)p;
      v8bf hi = *(const v8bf*)(p + 16);
#pragma unroll
      for (int i = 0; i < 8; ++i) { a[i] = lo[i]; a[8 + i] = hi[i]; }
    }
    v16bf br = load_b_bf16(whh, H_D, ncol,           k0, half);
    v16bf bz = load_b_bf16(whh, H_D, ncol + H_D,     k0, half);
    v16bf bn = load_b_bf16(whh, H_D, ncol + 2 * H_D, k0, half);
    ar = wmma_bf16(a, br, ar);
    az = wmma_bf16(a, bz, az);
    an = wmma_bf16(a, bn, an);
  }

  const float bhr = b_hh[ncol];
  const float bhz = b_hh[ncol + H_D];
  const float bhn = b_hh[ncol + 2 * H_D];
  const int mb = m0 + half * 8;
#pragma unroll
  for (int i = 0; i < 8; ++i) {
    const int b = mb + i;                         // batch index (M == B == 32)
    const size_t gi = (size_t)b * G3H;
    float r  = sigmoidf_(xw_t[gi + ncol]       + ar[i] + bhr);
    float zg = sigmoidf_(xw_t[gi + ncol + H_D] + az[i] + bhz);
    float nn = tanhf_(xw_t[gi + ncol + 2*H_D] + r * (an[i] + bhn));
    float hp = h_in[(size_t)b * H_D + ncol];
    float hn = (1.0f - zg) * nn + zg * hp;
    h_out[(size_t)b * H_D + ncol]  = hn;
    h_b_out[(size_t)b * H_D + ncol] = (__bf16)hn;
  }
}

// Heads: dense(concat(z,h)) -> mu / softplus(sigma) -> reparam -> planar flow.
// Tiny (D=Z=100): single workgroup, LDS staged, fp32 scalar.
__global__ void __launch_bounds__(1024)
heads_step(const float* __restrict__ h_new,       // [B,H] fp32
           float* __restrict__ zbuf,              // [B,Z] fp32 (in/out)
           const float* __restrict__ Wd,  const float* __restrict__ bd,
           const float* __restrict__ Wmu, const float* __restrict__ bmu,
           const float* __restrict__ Wsg, const float* __restrict__ bsg,
           const float* __restrict__ u1,
           const float* __restrict__ Wpf, const float* __restrict__ bpf,
           const float* __restrict__ noise_t,     // [B,Z] slice for step w
           float* __restrict__ out_z, float* __restrict__ out_mu,
           float* __restrict__ out_lv, int w)
{
  __shared__ float zsh [B_SZ * Z_D];
  __shared__ float dsh [B_SZ * D_D];
  __shared__ float zpre[B_SZ * Z_D];
  const int tid = threadIdx.x;

  for (int i = tid; i < B_SZ * Z_D; i += 1024) zsh[i] = zbuf[i];
  __syncthreads();

  // dense_out = concat(z_t, h_new) @ Wd^T + bd   (Wd: [D, Z+H])
  for (int idx = tid; idx < B_SZ * D_D; idx += 1024) {
    const int b = idx / D_D, d = idx - b * D_D;
    const float* wr = Wd + (size_t)d * (Z_D + H_D);
    const float* zr = zsh + b * Z_D;
    const float* hr = h_new + (size_t)b * H_D;
    float acc = bd[d];
    for (int k = 0; k < Z_D; ++k) acc += zr[k] * wr[k];
    const float* wh = wr + Z_D;
    for (int k = 0; k < H_D; ++k) acc += hr[k] * wh[k];
    dsh[idx] = acc;
  }
  __syncthreads();

  const float u = u1[0];
  for (int idx = tid; idx < B_SZ * Z_D; idx += 1024) {
    const int b = idx / Z_D, z = idx - b * Z_D;
    const float* dr = dsh + b * D_D;
    const float* wm = Wmu + (size_t)z * D_D;
    const float* ws = Wsg + (size_t)z * D_D;
    float am = bmu[z], as = bsg[z];
    for (int k = 0; k < D_D; ++k) { am += dr[k] * wm[k]; as += dr[k] * ws[k]; }
    float sp  = fmaxf(as, 0.0f) + log1pf(__expf(-fabsf(as)));   // softplus
    float eps = noise_t[b * Z_D + z];
    zpre[idx] = am + __expf(0.5f * sp) * eps;
    const size_t o = ((size_t)b * W_T + w) * Z_D + z;
    out_mu[o] = am;
    out_lv[o] = sp;
  }
  __syncthreads();

  for (int idx = tid; idx < B_SZ * Z_D; idx += 1024) {
    const int b = idx / Z_D, z = idx - b * Z_D;
    const float* zr = zpre + b * Z_D;
    const float* wp = Wpf + (size_t)z * Z_D;
    float acc = bpf[z];
    for (int k = 0; k < Z_D; ++k) acc += zr[k] * wp[k];
    float zn = zpre[idx] + u * tanhf_(acc);
    zbuf[idx] = zn;
    out_z[((size_t)b * W_T + w) * Z_D + z] = zn;
  }
}

// ---------------- launch ----------------

extern "C" void kernel_launch(void* const* d_in, const int* in_sizes, int n_in,
                              void* d_out, int out_size, void* d_ws, size_t ws_size,
                              hipStream_t stream)
{
  const float* x    = (const float*)d_in[0];
  const float* Wih  = (const float*)d_in[1];
  const float* Whh  = (const float*)d_in[2];
  const float* bih  = (const float*)d_in[3];
  const float* bhh  = (const float*)d_in[4];
  const float* Wd   = (const float*)d_in[5];
  const float* bd   = (const float*)d_in[6];
  const float* Wmu  = (const float*)d_in[7];
  const float* bmu  = (const float*)d_in[8];
  const float* Wsg  = (const float*)d_in[9];
  const float* bsg  = (const float*)d_in[10];
  const float* u    = (const float*)d_in[11];
  const float* Wpf  = (const float*)d_in[12];
  const float* bpf  = (const float*)d_in[13];
  const float* nz   = (const float*)d_in[14];

  char* ws = (char*)d_ws;
  size_t off = 0;
  auto take = [&](size_t bytes) -> void* {
    void* p = ws + off;
    off = (off + bytes + 255) & ~(size_t)255;
    return p;
  };
  __bf16* wih_b = (__bf16*)take((size_t)G3H * K_IN * 2);         // 25.2 MB
  __bf16* whh_b = (__bf16*)take((size_t)G3H * H_D * 2);          //  6.3 MB
  float*  xW    = (float*) take((size_t)W_T * B_SZ * G3H * 4);   // 100.7 MB
  float*  hF    = (float*) take((size_t)2 * B_SZ * H_D * 4);     // ping-pong
  __bf16* hB    = (__bf16*)take((size_t)2 * B_SZ * H_D * 2);
  float*  zbuf  = (float*) take((size_t)B_SZ * Z_D * 4);
  (void)ws_size; (void)in_sizes; (void)n_in; (void)out_size;

  float* out_z  = (float*)d_out;
  float* out_mu = out_z  + (size_t)B_SZ * W_T * Z_D;
  float* out_lv = out_mu + (size_t)B_SZ * W_T * Z_D;

  { int n2 = (G3H * K_IN) / 2;
    cvt_f32_to_bf16_k<<<(n2 + 255) / 256, 256, 0, stream>>>(Wih, (v2bf*)wih_b, n2); }
  { int n2 = (G3H * H_D) / 2;
    cvt_f32_to_bf16_k<<<(n2 + 255) / 256, 256, 0, stream>>>(Whh, (v2bf*)whh_b, n2); }
  init_state_k<<<(B_SZ * H_D + 255) / 256, 256, 0, stream>>>(hF, hB, zbuf);

  // 24576 wave-units / 8 waves per block = 3072 blocks
  input_proj_wmma<<<3072, 256, 0, stream>>>(x, wih_b, bih, xW);

  for (int w = 0; w < W_T; ++w) {
    const int cur = w & 1, nxt = cur ^ 1;
    gru_step_wmma<<<16, 256, 0, stream>>>(
        hB + (size_t)cur * B_SZ * H_D, whh_b, bhh,
        xW + (size_t)w * B_SZ * G3H,
        hF + (size_t)cur * B_SZ * H_D,
        hF + (size_t)nxt * B_SZ * H_D,
        hB + (size_t)nxt * B_SZ * H_D);
    heads_step<<<1, 1024, 0, stream>>>(
        hF + (size_t)nxt * B_SZ * H_D, zbuf,
        Wd, bd, Wmu, bmu, Wsg, bsg, u, Wpf, bpf,
        nz + (size_t)w * B_SZ * Z_D, out_z, out_mu, out_lv, w);
  }
}
